// StateSpaceLayer_17489106829858
// MI455X (gfx1250) — matrix operand order repaired and checked by reading, not compiled, not run on hardware
//
#include <hip/hip_runtime.h>
#include <hip/hip_bf16.h>

// ---------------- types -------------------------------------------------
typedef __attribute__((ext_vector_type(16))) __bf16 v16bf;
typedef __attribute__((ext_vector_type(2)))  __bf16 v2bf;
typedef __attribute__((ext_vector_type(8)))  float  v8f;
typedef __attribute__((ext_vector_type(4)))  unsigned int u32x4;

#define HID   1024
#define STATE 16
#define BATCH 4
#define SEQ   4096
#define CHUNK 64
#define NCHUNK (SEQ / CHUNK)          // 64 chunks per batch
#define NROWS  (BATCH * SEQ)          // 16384 token rows
#define MB     64                     // M rows per workgroup in main kernel
#define NWGS   (NROWS / MB)           // 256 workgroups

union FragBF { v16bf v; unsigned u[8]; };

// f32 pair -> packed bf16 dword (hardware convert)
__device__ __forceinline__ unsigned pack2(float lo, float hi) {
#if __has_builtin(__builtin_amdgcn_cvt_pk_bf16_f32)
  union { v2bf v; unsigned u; } c;
  c.v = __builtin_amdgcn_cvt_pk_bf16_f32(lo, hi);
  return c.u;
#else
  unsigned r;
  asm("v_cvt_pk_bf16_f32 %0, %1, %2" : "=v"(r) : "v"(lo), "v"(hi));
  return r;
#endif
}

// async 16-byte global->LDS copy (ASYNCcnt-tracked, GVS addressing)
__device__ __forceinline__ void async_copy16(unsigned lds_off, unsigned goff,
                                             unsigned long long base) {
  asm volatile("global_load_async_to_lds_b128 %0, %1, %2"
               :: "v"(lds_off), "v"(goff), "s"(base) : "memory");
}
__device__ __forceinline__ void wait_async0() {
  asm volatile("s_wait_asynccnt 0" ::: "memory");
}

// ---------------- 1) D (fp32) -> bf16 -----------------------------------
__global__ __launch_bounds__(256) void cvt_d_kernel(const float* __restrict__ D,
                                                    unsigned* __restrict__ Dbf) {
  int p = blockIdx.x * 256 + threadIdx.x;         // one bf16 pair per thread
  const float2 v = *(reinterpret_cast<const float2*>(D) + p);
  Dbf[p] = pack2(v.x, v.y);
}

// ---------------- 2) xB = x @ B.T via WMMA; also emits x_bf16 ------------
__global__ __launch_bounds__(128) void xb_kernel(const float* __restrict__ x,
                                                 const float* __restrict__ Bm,
                                                 float* __restrict__ xB,
                                                 unsigned* __restrict__ xbf) {
  int gwave = (blockIdx.x * 128 + threadIdx.x) >> 5;   // tile id 0..1023
  int lane  = threadIdx.x & 31;
  int lr = lane & 15, half = lane >> 4;
  size_t grow = (size_t)(gwave * 16 + lr);
  const float* xrow = x  + grow * HID;
  const float* brow = Bm + (size_t)lr * HID;           // state row lr
  unsigned* xbrow = xbf + grow * (HID / 2);            // bf16 row (dwords)
  v8f acc = {0.f,0.f,0.f,0.f,0.f,0.f,0.f,0.f};
  for (int kb = 0; kb < HID; kb += 32) {
    FragBF a, b;
    int ka0 = kb + half * 16;
    int ka1 = kb + half * 16 + 8;
#pragma unroll
    for (int d = 0; d < 4; ++d) a.u[d]     = pack2(xrow[ka0 + 2*d], xrow[ka0 + 2*d + 1]);
#pragma unroll
    for (int d = 0; d < 4; ++d) a.u[4 + d] = pack2(xrow[ka1 + 2*d], xrow[ka1 + 2*d + 1]);
    // persist converted x as bf16 (each lane covers disjoint 8-value spans)
    *(u32x4*)(xbrow + (ka0 >> 1)) = (u32x4){a.u[0], a.u[1], a.u[2], a.u[3]};
    *(u32x4*)(xbrow + (ka1 >> 1)) = (u32x4){a.u[4], a.u[5], a.u[6], a.u[7]};
    int kbb = kb + half * 16;
#pragma unroll
    for (int d = 0; d < 8; ++d) b.u[d] = pack2(brow[kbb + 2*d], brow[kbb + 2*d + 1]);
    acc = __builtin_amdgcn_wmma_f32_16x16x32_bf16(false, a.v, false, b.v,
                                                  (short)0, acc, false, false);
  }
#pragma unroll
  for (int r = 0; r < 8; ++r) {
    int M = r + half * 8;
    xB[(size_t)(gwave * 16 + M) * STATE + lr] = acc[r];
  }
}

// ---------------- 3) A powers A^0..A^64 ----------------------------------
__global__ __launch_bounds__(256) void apow_kernel(const float* __restrict__ A,
                                                   float* __restrict__ Apow) {
  __shared__ float prev[256];
  __shared__ float As[256];
  int t = threadIdx.x, r = t >> 4, c = t & 15;
  As[t] = A[t];
  float cur = (r == c) ? 1.f : 0.f;
  Apow[t] = cur;                                   // A^0 = I
  __syncthreads();
  for (int j = 1; j <= CHUNK; ++j) {
    prev[t] = cur;
    __syncthreads();
    float s = 0.f;
#pragma unroll
    for (int k = 0; k < 16; ++k) s += prev[r * 16 + k] * As[k * 16 + c];
    cur = s;
    Apow[j * 256 + t] = cur;
    __syncthreads();
  }
}

// ---------------- 4) per-chunk local scans (zero init) -------------------
__global__ __launch_bounds__(32) void local_scan_kernel(const float* __restrict__ xB,
                                                        const float* __restrict__ A,
                                                        float* __restrict__ slocal,
                                                        float* __restrict__ finals) {
  int chunk = blockIdx.x;                          // 0..255 = b*64 + c
  int lane = threadIdx.x, i = lane & 15;
  float Arow[16];
#pragma unroll
  for (int j = 0; j < 16; ++j) Arow[j] = A[i * 16 + j];
  float st = 0.f;
  size_t base = (size_t)chunk * CHUNK;
  for (int k = 0; k < CHUNK; ++k) {
    float s = xB[(base + k) * STATE + i];
#pragma unroll
    for (int j = 0; j < 16; ++j) s += Arow[j] * __shfl(st, j, 32);
    st = s;
    if (lane < 16) slocal[(base + k) * STATE + i] = st;
  }
  if (lane < 16) finals[chunk * STATE + i] = st;
}

// ---------------- 5) carry scan across chunks ----------------------------
__global__ __launch_bounds__(32) void carry_scan_kernel(const float* __restrict__ Apow,
                                                        const float* __restrict__ finals,
                                                        float* __restrict__ carries) {
  int b = blockIdx.x;
  int lane = threadIdx.x, i = lane & 15;
  const float* AL = Apow + CHUNK * 256;            // A^64
  float Arow[16];
#pragma unroll
  for (int j = 0; j < 16; ++j) Arow[j] = AL[i * 16 + j];
  float carry = 0.f;
  for (int c = 0; c < NCHUNK; ++c) {
    if (lane < 16) carries[(size_t)(b * NCHUNK + c) * STATE + i] = carry;
    float s = finals[(size_t)(b * NCHUNK + c) * STATE + i];
#pragma unroll
    for (int j = 0; j < 16; ++j) s += Arow[j] * __shfl(carry, j, 32);
    carry = s;
  }
}

// ---------------- 6) fix-up: state = A^{k+1} carry + local; emit bf16 ----
__global__ __launch_bounds__(256) void fixup_kernel(const float* __restrict__ slocal,
                                                    const float* __restrict__ Apow,
                                                    const float* __restrict__ carries,
                                                    unsigned short* __restrict__ states) {
  int t = blockIdx.x * 256 + threadIdx.x;          // 0..262143
  int i = t & 15;
  size_t row = (size_t)(t >> 4);
  int k = (int)(row & (CHUNK - 1));
  size_t cc = row >> 6;                            // b*64 + c
  const float* Ap = Apow + (size_t)(k + 1) * 256 + i * 16;
  const float* cr = carries + cc * STATE;
  float s = slocal[row * STATE + i];
#pragma unroll
  for (int j = 0; j < 16; ++j) s += Ap[j] * cr[j];
  states[row * STATE + i] = (unsigned short)(pack2(s, 0.f) & 0xffffu);
}

// ---------------- 7) fused: out = states@C.T + x@D.T, LayerNorm ----------
// 64 rows per WG, 32 waves. Each wave: 4 m-tiles x 2 n-tiles; one B fragment
// feeds 4 WMMAs (4x reuse). x tile arrives via async global->LDS copies.
#define XS_STRIDE_U32 520                          // 1040 bf16 per row (1024 + pad)
#define XS_STRIDE_B   2080
__global__ __launch_bounds__(1024) void main_kernel(const unsigned* __restrict__ xbf,
                                                    const unsigned* __restrict__ Dbf,
                                                    const float* __restrict__ Cm,
                                                    const unsigned short* __restrict__ states,
                                                    const float* __restrict__ gamma,
                                                    const float* __restrict__ beta,
                                                    float* __restrict__ out) {
  __shared__ __align__(16) unsigned xs[MB * XS_STRIDE_U32]; // 130 KB: x tile bf16
  __shared__ float rowsum[MB];
  __shared__ float rowsq[MB];
  __shared__ float muArr[MB];
  __shared__ float invArr[MB];

  int tid = threadIdx.x;
  int wave = tid >> 5, lane = tid & 31;
  int lr = lane & 15, half = lane >> 4;
  size_t blk = blockIdx.x;                         // rows [blk*64, blk*64+64)
  size_t row0 = blk * MB;

  if (tid < MB)          rowsum[tid] = 0.f;
  else if (tid < 2 * MB) rowsq[tid - MB] = 0.f;

  // Phase 1: async copy x_bf16 tile [64 x 2048B] into LDS (padded stride)
  {
    unsigned xs_base = (unsigned)(uintptr_t)(&xs[0]);
    unsigned long long gbase = (unsigned long long)(uintptr_t)(xbf) +
                               (unsigned long long)row0 * (HID * 2);
    for (int p = tid; p < MB * 128; p += 1024) {   // 16B chunks
      int r = p >> 7, c = p & 127;
      async_copy16(xs_base + r * XS_STRIDE_B + c * 16,
                   (unsigned)(r * (HID * 2) + c * 16), gbase);
    }
    wait_async0();
  }
  __syncthreads();

  // Phase 2: D-GEMM. Wave owns cols [wave*32, wave*32+32), all 64 rows.
  v8f acc[8];
#pragma unroll
  for (int t = 0; t < 8; ++t) acc[t] = (v8f){0.f,0.f,0.f,0.f,0.f,0.f,0.f,0.f};

  const unsigned* xr[4];
#pragma unroll
  for (int m = 0; m < 4; ++m) xr[m] = xs + (m * 16 + lr) * XS_STRIDE_U32;
  const unsigned* drow[2];
#pragma unroll
  for (int nt = 0; nt < 2; ++nt)
    drow[nt] = Dbf + (size_t)(wave * 32 + nt * 16 + lr) * (HID / 2);

  for (int kb = 0; kb < HID; kb += 32) {
    FragBF a[4];
    int doff = (kb >> 1) + half * 4;               // A frag: K kb+half*8 .. / +16
#pragma unroll
    for (int m = 0; m < 4; ++m) {
      u32x4 lo = *(const u32x4*)(xr[m] + doff);
      u32x4 hi = *(const u32x4*)(xr[m] + doff + 8);
#pragma unroll
      for (int d = 0; d < 4; ++d) { a[m].u[d] = lo[d]; a[m].u[4 + d] = hi[d]; }
    }
    int kdw = (kb >> 1) + half * 4;                // B frag: K kb+half*8 .. / +16
#pragma unroll
    for (int nt = 0; nt < 2; ++nt) {
      u32x4 b0 = *(const u32x4*)(drow[nt] + kdw);
      u32x4 b1 = *(const u32x4*)(drow[nt] + kdw + 8);
      FragBF b;
#pragma unroll
      for (int d = 0; d < 4; ++d) { b.u[d] = b0[d]; b.u[4 + d] = b1[d]; }
#pragma unroll
      for (int m = 0; m < 4; ++m)
        acc[(nt << 2) | m] = __builtin_amdgcn_wmma_f32_16x16x32_bf16(
            false, a[m].v, false, b.v, (short)0, acc[(nt << 2) | m], false, false);
    }
  }

  // Phase 3: + states @ C.T (K=16 zero-padded to 32); one B frag per n-tile.
  {
    FragBF sa[4];
#pragma unroll
    for (int m = 0; m < 4; ++m) {
      u32x4 sv = *(const u32x4*)(reinterpret_cast<const unsigned*>(states) +
                                 (row0 + m * 16 + lr) * (STATE / 2) + half * 4);
#pragma unroll
      for (int d = 0; d < 4; ++d) { sa[m].u[d] = sv[d]; sa[m].u[4 + d] = 0u; }
    }
#pragma unroll
    for (int nt = 0; nt < 2; ++nt) {
      int o = wave * 32 + nt * 16 + lr;
      FragBF cb;
      {                                            // K=half*8..+7 from C row o; K>=16 zero
        const float* crow = Cm + (size_t)o * STATE + half * 8;
#pragma unroll
        for (int d = 0; d < 4; ++d) { cb.u[d] = pack2(crow[2*d], crow[2*d + 1]); cb.u[4 + d] = 0u; }
      }
#pragma unroll
      for (int m = 0; m < 4; ++m)
        acc[(nt << 2) | m] = __builtin_amdgcn_wmma_f32_16x16x32_bf16(
            false, sa[m].v, false, cb.v, (short)0, acc[(nt << 2) | m], false, false);
    }
  }

  // Phase 4: per-row sum / sumsq partials (intra-half shfl reduce + LDS atomics)
#pragma unroll
  for (int t = 0; t < 8; ++t) {
    int m = t & 3;
#pragma unroll
    for (int r = 0; r < 8; ++r) {
      float v = acc[t][r];
      float s = v, q = v * v;
#pragma unroll
      for (int off = 1; off < 16; off <<= 1) {     // stays within 16-lane half
        s += __shfl_xor(s, off, 32);
        q += __shfl_xor(q, off, 32);
      }
      if (lr == 0) {                               // lane 0 (half 0) & lane 16 (half 1)
        int rw = m * 16 + r + half * 8;
        atomicAdd(&rowsum[rw], s);
        atomicAdd(&rowsq[rw], q);
      }
    }
  }
  __syncthreads();
  if (tid < MB) {
    float mu  = rowsum[tid] * (1.f / HID);
    float var = rowsq[tid] * (1.f / HID) - mu * mu;
    muArr[tid]  = mu;
    invArr[tid] = rsqrtf(var + 1e-5f);
  }
  __syncthreads();

  // Phase 5: normalize accumulators in place and store fp32
#pragma unroll
  for (int nt = 0; nt < 2; ++nt) {
    int col = wave * 32 + nt * 16 + lr;
    float g = gamma[col], bb = beta[col];
#pragma unroll
    for (int m = 0; m < 4; ++m) {
      int t = (nt << 2) | m;
#pragma unroll
      for (int r = 0; r < 8; ++r) {
        int rw = m * 16 + r + half * 8;
        float v = (acc[t][r] - muArr[rw]) * invArr[rw] * g + bb;
        out[(row0 + rw) * HID + col] = v;
      }
    }
  }
}

// ---------------- launcher ----------------------------------------------
extern "C" void kernel_launch(void* const* d_in, const int* in_sizes, int n_in,
                              void* d_out, int out_size, void* d_ws, size_t ws_size,
                              hipStream_t stream) {
  const float* x  = (const float*)d_in[0];
  const float* Am = (const float*)d_in[1];
  const float* Bm = (const float*)d_in[2];
  const float* Cm = (const float*)d_in[3];
  const float* Dm = (const float*)d_in[4];
  const float* gm = (const float*)d_in[5];
  const float* bt = (const float*)d_in[6];
  float* out = (float*)d_out;

  char* ws = (char*)d_ws;
  unsigned*       Dbf     = (unsigned*)(ws + 0);                        // 2 MB
  unsigned*       xbfw    = (unsigned*)(ws + 2097152);                  // 32 MB x_bf16
  float*          xBws    = (float*)(ws + 35651584);                    // 1 MB
  float*          slocal  = (float*)(ws + 36700160);                    // 1 MB
  float*          finals  = (float*)(ws + 37748736);                    // 16 KB
  float*          carries = (float*)(ws + 37765120);                    // 16 KB
  float*          Apow    = (float*)(ws + 37781504);                    // 65 KB
  unsigned short* statesb = (unsigned short*)(ws + 37848064);           // 512 KB

  cvt_d_kernel     <<<(HID * HID / 2) / 256, 256, 0, stream>>>(Dm, Dbf);
  xb_kernel        <<<(NROWS / 16) / 4, 128, 0, stream>>>(x, Bm, xBws, xbfw);
  apow_kernel      <<<1, 256, 0, stream>>>(Am, Apow);
  local_scan_kernel<<<BATCH * NCHUNK, 32, 0, stream>>>(xBws, Am, slocal, finals);
  carry_scan_kernel<<<BATCH, 32, 0, stream>>>(Apow, finals, carries);
  fixup_kernel     <<<(NROWS * STATE) / 256, 256, 0, stream>>>(slocal, Apow, carries, statesb);
  main_kernel      <<<NWGS, 1024, 0, stream>>>(xbfw, Dbf, Cm, statesb, gm, bt, out);
}
